// Model_79199196938718
// MI455X (gfx1250) — matrix operand order repaired and checked by reading, not compiled, not run on hardware
//
#include <hip/hip_runtime.h>
#include <math.h>

#define BDIM 16
#define LSEQ 1024
#define CDIM 64

typedef __attribute__((ext_vector_type(16))) _Float16 v16h;
typedef __attribute__((ext_vector_type(8)))  _Float16 h8;
typedef __attribute__((ext_vector_type(8)))  float    v8f;

__device__ __forceinline__ float sigmoidf_(float x) { return 1.0f / (1.0f + __expf(-x)); }

// Build 16-bit A fragment (16x32 layout) from one contiguous 16-float run.
__device__ __forceinline__ v16h mk_a_frag(float4 a0, float4 a1, float4 a2, float4 a3,
                                          float m) {
  v16h af;
  af[0]  = (_Float16)(a0.x * m); af[1]  = (_Float16)(a0.y * m);
  af[2]  = (_Float16)(a0.z * m); af[3]  = (_Float16)(a0.w * m);
  af[4]  = (_Float16)(a1.x * m); af[5]  = (_Float16)(a1.y * m);
  af[6]  = (_Float16)(a1.z * m); af[7]  = (_Float16)(a1.w * m);
  af[8]  = (_Float16)(a2.x * m); af[9]  = (_Float16)(a2.y * m);
  af[10] = (_Float16)(a2.z * m); af[11] = (_Float16)(a2.w * m);
  af[12] = (_Float16)(a3.x * m); af[13] = (_Float16)(a3.y * m);
  af[14] = (_Float16)(a3.z * m); af[15] = (_Float16)(a3.w * m);
  return af;
}

// Load 16 contiguous packed halves as a B fragment (32x16 layout, K-contig).
__device__ __forceinline__ v16h ld_b_frag(const _Float16* p) {
  const h8* hp = (const h8*)p;
  h8 lo = hp[0], hi = hp[1];
  v16h bf;
#pragma unroll
  for (int e = 0; e < 8; ++e) { bf[e] = lo[e]; bf[8 + e] = hi[e]; }
  return bf;
}

// ---------------------------------------------------------------------------
// Pack an arbitrary f32 weight view (element (k,n) = W[boff + k*sk + n*sn])
// into f16 laid out [N][K] (K contiguous).
// ---------------------------------------------------------------------------
__global__ void k_pack(const float* __restrict__ W, long long boff, int sk, int sn,
                       int K, int N, _Float16* __restrict__ out) {
  int i = blockIdx.x * 256 + threadIdx.x;
  if (i < K * N) {
    int n = i / K, k = i % K;
    out[(long long)n * K + k] =
        (_Float16)W[boff + (long long)k * sk + (long long)n * sn];
  }
}

// ---------------------------------------------------------------------------
// WMMA GEMM: Out = act( beta*Out + A @ Bp + bias ).  Bp is packed f16 [N][K].
// One wave per 16x64 output strip (4 WMMA tiles share one A fragment).
// Software-pipelined: loads for k-step i+1 are issued before WMMAs of step i.
// A element (m,k) = A[aoff + m*lda + k]; optional causal shift with zero-fill
// implemented branch-free (clamped address + mask multiply).
// M mult of 16, N mult of 64, K mult of 32.
// ---------------------------------------------------------------------------
__global__ void k_gemm(const float* __restrict__ A, int lda, long long aoff,
                       int a_shift, int lmod,
                       const _Float16* __restrict__ Bp,
                       const float* __restrict__ bias,
                       float* __restrict__ Out, int ldo,
                       int M, int N, int K, int beta, int act) {
  const int strips_n = N >> 6;
  const int strip = blockIdx.x;
  const int n0 = (strip % strips_n) << 6;
  const int m0 = (strip / strips_n) << 4;
  const int lane = threadIdx.x & 31;

  const int am = m0 + (lane & 15);
  float amask = 1.0f;
  long long arow;
  if (lmod > 0) {
    int l = am % lmod;
    int ok = (l >= a_shift);
    amask = ok ? 1.0f : 0.0f;
    int sr = ok ? (am - a_shift) : am;          // clamp to a valid row
    arow = aoff + (long long)sr * lda;
  } else {
    arow = aoff + (long long)am * lda;
  }
  const int akb = (lane < 16) ? 0 : 16;
  const int bn  = lane & 15;
  const int bkb = (lane < 16) ? 0 : 16;
  const _Float16* bcol0 = Bp + (long long)(n0 + bn) * K + bkb;

  // ---- prologue: fragments for k0 = 0
  const float* ap = A + arow + akb;
  v16h af = mk_a_frag(*(const float4*)(ap), *(const float4*)(ap + 4),
                      *(const float4*)(ap + 8), *(const float4*)(ap + 12), amask);
  v16h bf0 = ld_b_frag(bcol0);
  v16h bf1 = ld_b_frag(bcol0 + (long long)16 * K);
  v16h bf2 = ld_b_frag(bcol0 + (long long)32 * K);
  v16h bf3 = ld_b_frag(bcol0 + (long long)48 * K);

  v8f c0 = {}, c1 = {}, c2 = {}, c3 = {};
  int k0 = 0;
  for (; k0 + 32 < K; k0 += 32) {
    // issue next k-step's loads first (stay in flight behind the WMMAs)
    const float* apn = A + arow + (k0 + 32) + akb;
    float4 na0 = *(const float4*)(apn);
    float4 na1 = *(const float4*)(apn + 4);
    float4 na2 = *(const float4*)(apn + 8);
    float4 na3 = *(const float4*)(apn + 12);
    v16h nb0 = ld_b_frag(bcol0 + k0 + 32);
    v16h nb1 = ld_b_frag(bcol0 + (long long)16 * K + k0 + 32);
    v16h nb2 = ld_b_frag(bcol0 + (long long)32 * K + k0 + 32);
    v16h nb3 = ld_b_frag(bcol0 + (long long)48 * K + k0 + 32);
    // compute current k-step
    c0 = __builtin_amdgcn_wmma_f32_16x16x32_f16(false, af, false, bf0, (short)0, c0, false, false);
    c1 = __builtin_amdgcn_wmma_f32_16x16x32_f16(false, af, false, bf1, (short)0, c1, false, false);
    c2 = __builtin_amdgcn_wmma_f32_16x16x32_f16(false, af, false, bf2, (short)0, c2, false, false);
    c3 = __builtin_amdgcn_wmma_f32_16x16x32_f16(false, af, false, bf3, (short)0, c3, false, false);
    // rotate
    af = mk_a_frag(na0, na1, na2, na3, amask);
    bf0 = nb0; bf1 = nb1; bf2 = nb2; bf3 = nb3;
  }
  // epilogue k-step
  c0 = __builtin_amdgcn_wmma_f32_16x16x32_f16(false, af, false, bf0, (short)0, c0, false, false);
  c1 = __builtin_amdgcn_wmma_f32_16x16x32_f16(false, af, false, bf1, (short)0, c1, false, false);
  c2 = __builtin_amdgcn_wmma_f32_16x16x32_f16(false, af, false, bf2, (short)0, c2, false, false);
  c3 = __builtin_amdgcn_wmma_f32_16x16x32_f16(false, af, false, bf3, (short)0, c3, false, false);

  const int mo = m0 + ((lane < 16) ? 0 : 8);
#pragma unroll
  for (int tt = 0; tt < 4; ++tt) {
    v8f c = (tt == 0) ? c0 : (tt == 1) ? c1 : (tt == 2) ? c2 : c3;
    const int n = n0 + tt * 16 + (lane & 15);
    const float bv = bias ? bias[n] : 0.0f;
#pragma unroll
    for (int r = 0; r < 8; ++r) {
      long long idx = (long long)(mo + r) * ldo + n;
      float v = c[r] + bv;
      if (beta) v += Out[idx];
      if (act) v = fmaxf(v, 0.0f);
      Out[idx] = v;
    }
  }
}

// ---------------------------------------------------------------------------
// Hybrid multi-scale conv (k=1,3,5, 'same') + ReLU.  One block per (b,l), 64 thr.
// ---------------------------------------------------------------------------
__global__ void k_hyb(const float* __restrict__ x,
                      const float* __restrict__ w1, const float* __restrict__ b1,
                      const float* __restrict__ w3, const float* __restrict__ b3,
                      const float* __restrict__ w5, const float* __restrict__ b5,
                      float* __restrict__ y) {
  const int row = blockIdx.x;           // b*L + l
  const int l  = row & (LSEQ - 1);
  const int co = threadIdx.x;           // 0..63
  __shared__ float xr[5][CDIM];
#pragma unroll
  for (int p = 0; p < 5; ++p) {
    int ls = l - 2 + p;
    xr[p][co] = (ls >= 0 && ls < LSEQ) ? x[(long long)(row - 2 + p) * CDIM + co] : 0.0f;
  }
  __syncthreads();
  float acc = b1[co] + b3[co] + b5[co];
  for (int ci = 0; ci < CDIM; ++ci) {
    acc += xr[2][ci] * w1[co * 64 + ci];
    const float* w3p = w3 + (co * 64 + ci) * 3;
    acc += xr[1][ci] * w3p[0] + xr[2][ci] * w3p[1] + xr[3][ci] * w3p[2];
    const float* w5p = w5 + (co * 64 + ci) * 5;
    acc += xr[0][ci] * w5p[0] + xr[1][ci] * w5p[1] + xr[2][ci] * w5p[2]
         + xr[3][ci] * w5p[3] + xr[4][ci] * w5p[4];
  }
  y[(long long)row * CDIM + co] = fmaxf(acc, 0.0f);
}

// ---------------------------------------------------------------------------
// In-place residual + LayerNorm over last dim (64).  One block per row.
// ---------------------------------------------------------------------------
__global__ void k_add_ln(float* __restrict__ y, const float* __restrict__ a,
                         const float* __restrict__ g, const float* __restrict__ be) {
  const int row = blockIdx.x;
  const int t = threadIdx.x;            // 64
  __shared__ float s1[64], s2[64];
  float v = y[(long long)row * 64 + t] + a[(long long)row * 64 + t];
  s1[t] = v; s2[t] = v * v;
  __syncthreads();
  for (int o = 32; o > 0; o >>= 1) {
    if (t < o) { s1[t] += s1[t + o]; s2[t] += s2[t + o]; }
    __syncthreads();
  }
  float mean = s1[0] * (1.0f / 64.0f);
  float var  = s2[0] * (1.0f / 64.0f) - mean * mean;
  y[(long long)row * 64 + t] = (v - mean) * rsqrtf(var + 1e-5f) * g[t] + be[t];
}

// ---------------------------------------------------------------------------
// Informer full softmax attention, h=4, dh=16.  One block per (b,h,q), 128 thr.
// ---------------------------------------------------------------------------
__global__ void k_inf_attn(const float* __restrict__ Q, const float* __restrict__ Kt,
                           const float* __restrict__ V, float* __restrict__ ctx) {
  const int q = blockIdx.x & (LSEQ - 1);
  const int h = (blockIdx.x >> 10) & 3;
  const int b = blockIdx.x >> 12;
  const int t = threadIdx.x;            // 128
  __shared__ float qv[16];
  __shared__ float s[LSEQ];
  __shared__ float red[128];
  __shared__ float part[128 * 16];
  if (t < 16) qv[t] = Q[((long long)(b * LSEQ + q)) * 64 + h * 16 + t];
  __syncthreads();
  float4 q0 = ((const float4*)qv)[0], q1 = ((const float4*)qv)[1];
  float4 q2 = ((const float4*)qv)[2], q3 = ((const float4*)qv)[3];
  float lmax = -1e30f;
  for (int kk = t; kk < LSEQ; kk += 128) {
    const float4* kp = (const float4*)(Kt + ((long long)(b * LSEQ + kk)) * 64 + h * 16);
    float4 k0 = kp[0], k1 = kp[1], k2 = kp[2], k3 = kp[3];
    float d = q0.x*k0.x + q0.y*k0.y + q0.z*k0.z + q0.w*k0.w
            + q1.x*k1.x + q1.y*k1.y + q1.z*k1.z + q1.w*k1.w
            + q2.x*k2.x + q2.y*k2.y + q2.z*k2.z + q2.w*k2.w
            + q3.x*k3.x + q3.y*k3.y + q3.z*k3.z + q3.w*k3.w;
    d *= 0.25f;                          // 1/sqrt(16)
    s[kk] = d;
    lmax = fmaxf(lmax, d);
  }
  red[t] = lmax; __syncthreads();
  for (int o = 64; o > 0; o >>= 1) { if (t < o) red[t] = fmaxf(red[t], red[t + o]); __syncthreads(); }
  const float mx = red[0];
  __syncthreads();
  float lsum = 0.0f;
  float4 p0 = {0,0,0,0}, p1 = {0,0,0,0}, p2 = {0,0,0,0}, p3 = {0,0,0,0};
  for (int kk = t; kk < LSEQ; kk += 128) {
    float ev = __expf(s[kk] - mx);
    lsum += ev;
    const float4* vp = (const float4*)(V + ((long long)(b * LSEQ + kk)) * 64 + h * 16);
    float4 v0 = vp[0], v1 = vp[1], v2 = vp[2], v3 = vp[3];
    p0.x += ev*v0.x; p0.y += ev*v0.y; p0.z += ev*v0.z; p0.w += ev*v0.w;
    p1.x += ev*v1.x; p1.y += ev*v1.y; p1.z += ev*v1.z; p1.w += ev*v1.w;
    p2.x += ev*v2.x; p2.y += ev*v2.y; p2.z += ev*v2.z; p2.w += ev*v2.w;
    p3.x += ev*v3.x; p3.y += ev*v3.y; p3.z += ev*v3.z; p3.w += ev*v3.w;
  }
  red[t] = lsum; __syncthreads();
  for (int o = 64; o > 0; o >>= 1) { if (t < o) red[t] += red[t + o]; __syncthreads(); }
  const float inv = 1.0f / red[0];
  float4* pr = (float4*)(part + t * 16);
  pr[0] = p0; pr[1] = p1; pr[2] = p2; pr[3] = p3;
  __syncthreads();
  if (t < 16) {
    float acc = 0.0f;
    for (int w = 0; w < 128; ++w) acc += part[w * 16 + t];
    ctx[((long long)(b * LSEQ + q)) * 64 + h * 16 + t] = acc * inv;
  }
}

// out[i] = relu(a[i] + b[i])
__global__ void k_add_relu(const float* __restrict__ a, const float* __restrict__ b,
                           float* __restrict__ o, long long n) {
  long long i = (long long)blockIdx.x * blockDim.x + threadIdx.x;
  if (i < n) o[i] = fmaxf(a[i] + b[i], 0.0f);
}

// ---------------------------------------------------------------------------
// Bidirectional GRU scan: 2 blocks (dir), 768 threads (24 waves).
// gh = h @ Whh^T via WMMA with packed f16 Whh ([768][256]); h, gh in LDS.
// ---------------------------------------------------------------------------
__global__ void __launch_bounds__(768) k_gru_scan(
    const float* __restrict__ giF, const float* __restrict__ giB,
    const _Float16* __restrict__ WhhFP, const _Float16* __restrict__ WhhBP,
    const float* __restrict__ bhhF, const float* __restrict__ bhhB,
    float* __restrict__ go) {
  const int dir = blockIdx.x;           // 0 fwd, 1 bwd
  const float* gi        = dir ? giB   : giF;
  const _Float16* WhhP   = dir ? WhhBP : WhhFP;
  const float* bhh       = dir ? bhhB  : bhhF;
  const int col0 = dir * 256;
  const int tid  = threadIdx.x;
  const int wave = tid >> 5;
  const int lane = tid & 31;
  __shared__ float h_s[16][256];        // 16 KB
  __shared__ float gh_s[16][768];       // 48 KB
  for (int i = tid; i < 16 * 256; i += 768) h_s[i >> 8][i & 255] = 0.0f;
  __syncthreads();
  const int akb = (lane < 16) ? 0 : 16;
  const int bkb = (lane < 16) ? 0 : 16;
  const int mrow = lane & 15;
  for (int step = 0; step < LSEQ; ++step) {
    const int l = dir ? (LSEQ - 1 - step) : step;
#pragma unroll
    for (int tt = 0; tt < 2; ++tt) {
      const int n0 = (wave * 2 + tt) << 4;
      const int n  = n0 + (lane & 15);
      const _Float16* bcol = WhhP + (long long)n * 256 + bkb;
      v8f c = {};
#pragma unroll
      for (int k0 = 0; k0 < 256; k0 += 32) {
        const float* hp = &h_s[mrow][k0 + akb];
        float4 a0 = *(const float4*)(hp);
        float4 a1 = *(const float4*)(hp + 4);
        float4 a2 = *(const float4*)(hp + 8);
        float4 a3 = *(const float4*)(hp + 12);
        v16h af = mk_a_frag(a0, a1, a2, a3, 1.0f);
        v16h bf = ld_b_frag(bcol + k0);
        c = __builtin_amdgcn_wmma_f32_16x16x32_f16(false, af, false, bf, (short)0, c,
                                                   false, false);
      }
      const int mo = (lane < 16) ? 0 : 8;
#pragma unroll
      for (int r = 0; r < 8; ++r) gh_s[mo + r][n] = c[r];
    }
    __syncthreads();
    for (int idx = tid; idx < 16 * 256; idx += 768) {
      const int b = idx >> 8;
      const int j = idx & 255;
      const long long gbase = ((long long)b * LSEQ + l) * 768 + j;
      float r  = sigmoidf_(gi[gbase]       + gh_s[b][j]       + bhh[j]);
      float z  = sigmoidf_(gi[gbase + 256] + gh_s[b][j + 256] + bhh[j + 256]);
      float nn = tanhf(gi[gbase + 512] + r * (gh_s[b][j + 512] + bhh[j + 512]));
      float hp = h_s[b][j];
      float hn = (1.0f - z) * nn + z * hp;
      h_s[b][j] = hn;
      go[((long long)b * LSEQ + l) * 512 + col0 + j] = hn;
    }
    __syncthreads();
  }
}

// ---------------------------------------------------------------------------
// Sparse attention for the LAST query only.  One block per (b,h), 256 threads.
// Exact k-th-largest (k=102) threshold via 32-bit MSB radix select.
// ---------------------------------------------------------------------------
__global__ void k_sparse_attn(const float* __restrict__ QL, const float* __restrict__ Kb,
                              const float* __restrict__ Vb, float* __restrict__ ctx) {
  const int h = blockIdx.x & 1;
  const int b = blockIdx.x >> 1;
  const int t = threadIdx.x;            // 256
  __shared__ float qv[256];
  __shared__ float s[LSEQ];
  __shared__ unsigned int u[LSEQ];
  __shared__ float red[256];
  __shared__ int cnt;
  qv[t] = QL[(long long)b * 512 + h * 256 + t];
  __syncthreads();
  for (int kk = t; kk < LSEQ; kk += 256) {
    const float4* kp = (const float4*)(Kb + ((long long)(b * LSEQ + kk)) * 512 + h * 256);
    const float4* q4 = (const float4*)qv;
    float d = 0.0f;
    for (int e = 0; e < 64; ++e) {
      float4 kv = kp[e];
      float4 qq = q4[e];
      d += qq.x*kv.x + qq.y*kv.y + qq.z*kv.z + qq.w*kv.w;
    }
    s[kk] = d * 0.0625f;                 // 1/sqrt(256)
  }
  __syncthreads();
  for (int kk = t; kk < LSEQ; kk += 256) {
    unsigned int bits = __float_as_uint(s[kk]);
    u[kk] = (bits & 0x80000000u) ? ~bits : (bits | 0x80000000u);
  }
  __syncthreads();
  unsigned int prefix = 0u;
  int kneed = 102;                       // max(1, int(1024*(1-0.9)))
  for (int bit = 31; bit >= 0; --bit) {
    if (t == 0) cnt = 0;
    __syncthreads();
    const unsigned int bmask = 1u << bit;
    const unsigned int hmask = (bit == 31) ? 0u : (~0u << (bit + 1));
    int local = 0;
    for (int kk = t; kk < LSEQ; kk += 256) {
      unsigned int uv = u[kk];
      if (((uv ^ prefix) & hmask) == 0u && (uv & bmask)) local++;
    }
    atomicAdd(&cnt, local);
    __syncthreads();
    int cval = cnt;
    __syncthreads();
    if (cval >= kneed) prefix |= bmask; else kneed -= cval;
  }
  float thr = (prefix & 0x80000000u) ? __uint_as_float(prefix & 0x7FFFFFFFu)
                                     : __uint_as_float(~prefix);
  float lmax = -1e30f;
  for (int kk = t; kk < LSEQ; kk += 256) lmax = fmaxf(lmax, s[kk]);
  red[t] = lmax; __syncthreads();
  for (int o = 128; o > 0; o >>= 1) { if (t < o) red[t] = fmaxf(red[t], red[t + o]); __syncthreads(); }
  const float mx = red[0];
  __syncthreads();
  float lsum = 0.0f;
  for (int kk = t; kk < LSEQ; kk += 256) {
    float e = (s[kk] >= thr) ? __expf(s[kk] - mx) : 0.0f;
    s[kk] = e;
    lsum += e;
  }
  __syncthreads();
  red[t] = lsum; __syncthreads();
  for (int o = 128; o > 0; o >>= 1) { if (t < o) red[t] += red[t + o]; __syncthreads(); }
  const float inv = 1.0f / red[0];
  __syncthreads();
  float acc = 0.0f;
  for (int kk = 0; kk < LSEQ; ++kk)
    acc += s[kk] * Vb[((long long)(b * LSEQ + kk)) * 512 + h * 256 + t];
  ctx[(long long)b * 512 + h * 256 + t] = acc * inv;
}

// out[b] = att[b,:] . fc_W + fc_b
__global__ void k_head(const float* __restrict__ att, const float* __restrict__ fw,
                       const float* __restrict__ fb, float* __restrict__ out) {
  const int b = blockIdx.x;
  const int t = threadIdx.x;            // 64
  __shared__ float red[64];
  float acc = 0.0f;
  for (int d = t; d < 512; d += 64) acc += att[(long long)b * 512 + d] * fw[d];
  red[t] = acc; __syncthreads();
  for (int o = 32; o > 0; o >>= 1) { if (t < o) red[t] += red[t + o]; __syncthreads(); }
  if (t == 0) out[b] = red[0] + fb[0];
}

// ---------------------------------------------------------------------------
extern "C" void kernel_launch(void* const* d_in, const int* in_sizes, int n_in,
                              void* d_out, int out_size, void* d_ws, size_t ws_size,
                              hipStream_t stream) {
  (void)in_sizes; (void)out_size; (void)ws_size;
  const float* P[53];
  for (int i = 0; i < 53 && i < n_in; ++i) P[i] = (const float*)d_in[i];
  const float* x    = P[0];
  const float* hw1  = P[1];  const float* hb1  = P[2];
  const float* hw3  = P[3];  const float* hb3  = P[4];
  const float* hw5  = P[5];  const float* hb5  = P[6];
  const float* iWq  = P[7];  const float* iWk  = P[8];
  const float* iWv  = P[9];  const float* iWo  = P[10];
  const float* iW1  = P[11]; const float* ib1  = P[12];
  const float* iW2  = P[13]; const float* ib2  = P[14];
  const float* ig1  = P[15]; const float* ibe1 = P[16];
  const float* ig2  = P[17]; const float* ibe2 = P[18];
  const float* t0w1 = P[19]; const float* t0b1 = P[20];
  const float* t0w2 = P[21]; const float* t0b2 = P[22];
  const float* t1w1 = P[23]; const float* t1b1 = P[24];
  const float* t1w2 = P[25]; const float* t1b2 = P[26];
  const float* t1wd = P[27]; const float* t1bd = P[28];
  const float* t2w1 = P[29]; const float* t2b1 = P[30];
  const float* t2w2 = P[31]; const float* t2b2 = P[32];
  const float* t2wd = P[33]; const float* t2bd = P[34];
  const float* gWihF = P[35]; const float* gWhhF = P[36];
  const float* gbihF = P[37]; const float* gbhhF = P[38];
  const float* gWihB = P[39]; const float* gWhhB = P[40];
  const float* gbihB = P[41]; const float* gbhhB = P[42];
  const float* aWq  = P[43]; const float* abq  = P[44];
  const float* aWk  = P[45]; const float* abk  = P[46];
  const float* aWv  = P[47]; const float* abv  = P[48];
  const float* aWo  = P[49]; const float* abo  = P[50];
  const float* fcW  = P[51]; const float* fcb  = P[52];

  const int ML = BDIM * LSEQ;                 // 16384 rows
  const long long S64  = (long long)ML * 64;
  const long long S256 = (long long)ML * 256;
  const long long S512 = (long long)ML * 512;
  const long long S768 = (long long)ML * 768;

  float* ws  = (float*)d_ws;
  float* A   = ws;               // [ML,64]  y / y1 / y2
  float* BQ  = A   + S64;
  float* BK  = BQ  + S64;
  float* BV  = BK  + S64;
  float* BC  = BV  + S64;        // informer ctx
  float* BB  = BC  + S64;        // tmp 64ch
  float* F   = BB  + S64;        // [ML,256] ffn hidden / TCN o1
  float* T1  = F   + S256;
  float* T2  = T1  + S256;
  float* T3  = T2  + S256;
  float* T0  = T3  + S256;       // final = g
  float* GIF = T0  + S256;       // [ML,768]
  float* GIB = GIF + S768;
  float* GO  = GIB + S768;       // [ML,512]
  float* QL  = GO  + S512;       // [16,512]
  float* CTX = QL  + 16 * 512;
  float* ATT = CTX + 16 * 512;
  float* KB  = GIF;              // reuse (gi dead after scan)
  float* VB  = GIB;

  // packed-f16 weight arena
  _Float16* pk = (_Float16*)(ATT + 16 * 512);
  long long po = 0;
  auto palloc = [&](long long sz) { _Float16* p = pk + po; po += sz; return p; };
  _Float16* iWqP   = palloc(64 * 64);
  _Float16* iWkP   = palloc(64 * 64);
  _Float16* iWvP   = palloc(64 * 64);
  _Float16* iWoP   = palloc(64 * 64);
  _Float16* iW1P   = palloc(64 * 256);
  _Float16* iW2P   = palloc(256 * 64);
  _Float16* t0w1aP = palloc(64 * 64);   _Float16* t0w1bP = palloc(64 * 64);
  _Float16* t0w2aP = palloc(64 * 64);   _Float16* t0w2bP = palloc(64 * 64);
  _Float16* t1w1aP = palloc(64 * 128);  _Float16* t1w1bP = palloc(64 * 128);
  _Float16* t1w2aP = palloc(128 * 128); _Float16* t1w2bP = palloc(128 * 128);
  _Float16* t1wdP  = palloc(64 * 128);
  _Float16* t2w1aP = palloc(128 * 256); _Float16* t2w1bP = palloc(128 * 256);
  _Float16* t2w2aP = palloc(256 * 256); _Float16* t2w2bP = palloc(256 * 256);
  _Float16* t2wdP  = palloc(128 * 256);
  _Float16* gWihFP = palloc(256 * 768); _Float16* gWihBP = palloc(256 * 768);
  _Float16* gWhhFP = palloc(256 * 768); _Float16* gWhhBP = palloc(256 * 768);
  _Float16* aWqP   = palloc(512 * 512); _Float16* aWkP   = palloc(512 * 512);
  _Float16* aWvP   = palloc(512 * 512); _Float16* aWoP   = palloc(512 * 512);

  auto pack = [&](const float* W, long long boff, int sk, int sn, int K, int N,
                  _Float16* out) {
    int total = K * N;
    k_pack<<<(total + 255) / 256, 256, 0, stream>>>(W, boff, sk, sn, K, N, out);
  };
  // dense [K,N] weights: sk=N, sn=1.  transposed ([N,K] storage): sk=1, sn=K.
  pack(iWq, 0, 64, 1, 64, 64, iWqP);
  pack(iWk, 0, 64, 1, 64, 64, iWkP);
  pack(iWv, 0, 64, 1, 64, 64, iWvP);
  pack(iWo, 0, 64, 1, 64, 64, iWoP);
  pack(iW1, 0, 256, 1, 64, 256, iW1P);
  pack(iW2, 0, 64, 1, 256, 64, iW2P);
  pack(t0w1, 0, 2, 128, 64, 64, t0w1aP);  pack(t0w1, 1, 2, 128, 64, 64, t0w1bP);
  pack(t0w2, 0, 2, 128, 64, 64, t0w2aP);  pack(t0w2, 1, 2, 128, 64, 64, t0w2bP);
  pack(t1w1, 0, 2, 128, 64, 128, t1w1aP); pack(t1w1, 1, 2, 128, 64, 128, t1w1bP);
  pack(t1w2, 0, 2, 256, 128, 128, t1w2aP);pack(t1w2, 1, 2, 256, 128, 128, t1w2bP);
  pack(t1wd, 0, 1, 64, 64, 128, t1wdP);
  pack(t2w1, 0, 2, 256, 128, 256, t2w1aP);pack(t2w1, 1, 2, 256, 128, 256, t2w1bP);
  pack(t2w2, 0, 2, 512, 256, 256, t2w2aP);pack(t2w2, 1, 2, 512, 256, 256, t2w2bP);
  pack(t2wd, 0, 1, 128, 128, 256, t2wdP);
  pack(gWihF, 0, 1, 256, 256, 768, gWihFP);
  pack(gWihB, 0, 1, 256, 256, 768, gWihBP);
  pack(gWhhF, 0, 1, 256, 256, 768, gWhhFP);
  pack(gWhhB, 0, 1, 256, 256, 768, gWhhBP);
  pack(aWq, 0, 512, 1, 512, 512, aWqP);
  pack(aWk, 0, 512, 1, 512, 512, aWkP);
  pack(aWv, 0, 512, 1, 512, 512, aWvP);
  pack(aWo, 0, 512, 1, 512, 512, aWoP);

  auto gemm = [&](const float* Am, int lda, long long aoff, int ashift, int lmod,
                  const _Float16* Bp, const float* bias,
                  float* Out, int ldo, int M, int N, int K, int beta, int act) {
    dim3 grid((unsigned)((M / 16) * (N / 64)));
    k_gemm<<<grid, 32, 0, stream>>>(Am, lda, aoff, ashift, lmod, Bp, bias,
                                    Out, ldo, M, N, K, beta, act);
  };

  // --- hybrid multi-scale conv + relu -> A
  k_hyb<<<ML, 64, 0, stream>>>(x, hw1, hb1, hw3, hb3, hw5, hb5, A);
  // --- Informer self-attention (no biases in mha)
  gemm(A, 64, 0, 0, 0, iWqP, nullptr, BQ, 64, ML, 64, 64, 0, 0);
  gemm(A, 64, 0, 0, 0, iWkP, nullptr, BK, 64, ML, 64, 64, 0, 0);
  gemm(A, 64, 0, 0, 0, iWvP, nullptr, BV, 64, ML, 64, 64, 0, 0);
  k_inf_attn<<<BDIM * 4 * LSEQ, 128, 0, stream>>>(BQ, BK, BV, BC);
  gemm(BC, 64, 0, 0, 0, iWoP, nullptr, BB, 64, ML, 64, 64, 0, 0);
  k_add_ln<<<ML, 64, 0, stream>>>(A, BB, ig1, ibe1);
  // --- FFN
  gemm(A, 64, 0, 0, 0, iW1P, ib1, F, 256, ML, 256, 64, 0, 1);
  gemm(F, 256, 0, 0, 0, iW2P, ib2, BB, 64, ML, 64, 256, 0, 0);
  k_add_ln<<<ML, 64, 0, stream>>>(A, BB, ig2, ibe2);
  // --- TCN block 0 (64->64, dil=1), tin=A
  gemm(A, 64, 0, 1, LSEQ, t0w1aP, t0b1,   F,  64, ML, 64, 64, 0, 0);
  gemm(A, 64, 0, 0, 0,    t0w1bP, nullptr, F,  64, ML, 64, 64, 1, 1);
  gemm(F, 64, 0, 1, LSEQ, t0w2aP, t0b2,   T1, 64, ML, 64, 64, 0, 0);
  gemm(F, 64, 0, 0, 0,    t0w2bP, nullptr, T1, 64, ML, 64, 64, 1, 1);
  k_add_relu<<<(unsigned)((S64 + 255) / 256), 256, 0, stream>>>(T1, A, T0, S64);
  // --- TCN block 1 (64->128, dil=2), tin=T0
  gemm(T0, 64, 0, 2, LSEQ, t1w1aP, t1b1,   F,  128, ML, 128, 64, 0, 0);
  gemm(T0, 64, 0, 0, 0,    t1w1bP, nullptr, F,  128, ML, 128, 64, 1, 1);
  gemm(F, 128, 0, 2, LSEQ, t1w2aP, t1b2,   T1, 128, ML, 128, 128, 0, 0);
  gemm(F, 128, 0, 0, 0,    t1w2bP, nullptr, T1, 128, ML, 128, 128, 1, 1);
  gemm(T0, 64, 0, 0, 0,    t1wdP,  t1bd,   T3, 128, ML, 128, 64, 0, 0);
  k_add_relu<<<(unsigned)(((long long)ML * 128 + 255) / 256), 256, 0, stream>>>(
      T1, T3, T2, (long long)ML * 128);
  // --- TCN block 2 (128->256, dil=4), tin=T2
  gemm(T2, 128, 0, 4, LSEQ, t2w1aP, t2b1,   F,  256, ML, 256, 128, 0, 0);
  gemm(T2, 128, 0, 0, 0,    t2w1bP, nullptr, F,  256, ML, 256, 128, 1, 1);
  gemm(F, 256, 0, 4, LSEQ,  t2w2aP, t2b2,   T1, 256, ML, 256, 256, 0, 0);
  gemm(F, 256, 0, 0, 0,     t2w2bP, nullptr, T1, 256, ML, 256, 256, 1, 1);
  gemm(T2, 128, 0, 0, 0,    t2wdP,  t2bd,   T3, 256, ML, 256, 128, 0, 0);
  k_add_relu<<<(unsigned)((S256 + 255) / 256), 256, 0, stream>>>(T1, T3, T0, S256);
  // --- GRU input gates (gi = g @ Wih^T + bih)
  gemm(T0, 256, 0, 0, 0, gWihFP, gbihF, GIF, 768, ML, 768, 256, 0, 0);
  gemm(T0, 256, 0, 0, 0, gWihBP, gbihB, GIB, 768, ML, 768, 256, 0, 0);
  k_gru_scan<<<2, 768, 0, stream>>>(GIF, GIB, gWhhFP, gWhhBP, gbhhF, gbhhB, GO);
  // --- sparse attention: K,V full; Q only for last timestep
  gemm(GO, 512, 0, 0, 0, aWkP, abk, KB, 512, ML, 512, 512, 0, 0);
  gemm(GO, 512, 0, 0, 0, aWvP, abv, VB, 512, ML, 512, 512, 0, 0);
  gemm(GO, LSEQ * 512, (long long)(LSEQ - 1) * 512, 0, 0, aWqP, abq,
       QL, 512, BDIM, 512, 512, 0, 0);
  k_sparse_attn<<<BDIM * 2, 256, 0, stream>>>(QL, KB, VB, CTX);
  gemm(CTX, 512, 0, 0, 0, aWoP, abo, ATT, 512, BDIM, 512, 512, 0, 0);
  k_head<<<BDIM, 64, 0, stream>>>(ATT, fcW, fcb, (float*)d_out);
}
